// SparseSphereConv_31731218383017
// MI455X (gfx1250) — matrix operand order repaired and checked by reading, not compiled, not run on hardware
//
#include <hip/hip_runtime.h>

typedef _Float16 half_t;
typedef __attribute__((ext_vector_type(16))) _Float16 v16h;
typedef __attribute__((ext_vector_type(8)))  _Float16 v8h;
typedef __attribute__((ext_vector_type(8)))  float    v8f;

#define BS     4
#define C_IN   32
#define C_OUT  32
#define NVERT  40962
#define KK     9
#define KD     (C_IN*KK)     // 288 contraction depth
#define TILE_V 16
#define M_ROWS (BS*TILE_V)   // 64 GEMM rows per block
#define ASTRIDE 296          // f16 elems; 592B row stride (16B aligned)

// ---- prep: conv_w f32 [C_OUT, C_IN, K] -> f16, same flat layout ----
// Row o of this array (contiguous over (c,k)) is exactly the WMMA B-fragment
// source row for output channel o.
__global__ void wcvt_kernel(const float* __restrict__ w, half_t* __restrict__ wh) {
    int i = blockIdx.x * 256 + threadIdx.x;
    if (i < C_OUT * KD) wh[i] = (half_t)w[i];
}

__global__ __launch_bounds__(256)
void sphereconv_kernel(const float* __restrict__ x,
                       const int*   __restrict__ nbr,
                       const float* __restrict__ itp,
                       const half_t* __restrict__ wh,
                       const float* __restrict__ bias,
                       float* __restrict__ out) {
    __shared__ __align__(16) half_t As[M_ROWS * ASTRIDE];   // f16 A tile [64][288]
    __shared__ float sitp[TILE_V * KK * KK];
    __shared__ int   sidx[TILE_V * KK];
    __shared__ int   smask[M_ROWS];

    const int tid = threadIdx.x;
    const int v0  = blockIdx.x * TILE_V;

    // ---- stage per-vertex metadata ----
    for (int i = tid; i < TILE_V * KK; i += 256) {
        int vt = i / KK;
        int v  = v0 + vt;
        sidx[i] = (v < NVERT) ? nbr[(size_t)v * KK + (i % KK)] : 0;
    }
    for (int i = tid; i < TILE_V * KK * KK; i += 256) {
        int vt = i / (KK * KK);
        int v  = v0 + vt;
        sitp[i] = (v < NVERT) ? itp[(size_t)v * KK * KK + (i % (KK * KK))] : 0.0f;
    }
    if (tid < M_ROWS) smask[tid] = 0;
    __syncthreads();

    // ---- gather + 9x9 interpolation (VALU), build f16 A tile ----
    // task = (vt, b, c): 16*4*32 = 2048 tasks, 8 per thread
    for (int it = 0; it < 8; ++it) {
        int task = tid + it * 256;
        int c  = task & 31;
        int b  = (task >> 5) & 3;
        int vt = task >> 7;
        const float* xp = x + (size_t)(b * C_IN + c) * NVERT;
        float g[KK];
        int nz = 0;
        #pragma unroll
        for (int k = 0; k < KK; ++k) {
            float gv = xp[sidx[vt * KK + k]];     // L2-resident gather
            g[k] = gv;
            nz |= (gv != 0.0f);
        }
        const float* ip = &sitp[vt * KK * KK];
        int row = b * TILE_V + vt;
        half_t* ap = &As[row * ASTRIDE + c * KK];
        #pragma unroll
        for (int j = 0; j < KK; ++j) {
            float acc = 0.0f;
            #pragma unroll
            for (int k = 0; k < KK; ++k) acc = fmaf(g[k], ip[k * KK + j], acc);
            ap[j] = (half_t)acc;
        }
        if (nz) atomicOr(&smask[row], 1);
    }
    __syncthreads();

    // ---- WMMA phase: 8 waves, wave -> (b = w&3 : 16 vertex rows, nb = w>>2 : 16 out channels)
    const int lane     = tid & 31;
    const int wv       = tid >> 5;
    const int b        = wv & 3;
    const int nb       = wv >> 2;
    const int half_sel = lane >> 4;   // 0: lanes 0-15, 1: lanes 16-31
    const int ln16     = lane & 15;

    const half_t* arow = &As[(b * TILE_V + ln16) * ASTRIDE];          // M = ln16
    const half_t* brow = wh + (size_t)(nb * 16 + ln16) * KD;          // N = ln16

    v8f acc = {};
    #pragma unroll
    for (int kb = 0; kb < KD / 32; ++kb) {                            // 9 K-steps of 32
        // A frag (16-bit 16x32 layout): lanes0-15 K 0..15, lanes16-31 K 16..31 (contiguous, matches B)
        int klo = kb * 32 + half_sel * 16;
        v8h alo = *(const v8h*)(arow + klo);
        v8h ahi = *(const v8h*)(arow + klo + 8);
        v16h a = __builtin_shufflevector(alo, ahi, 0,1,2,3,4,5,6,7,8,9,10,11,12,13,14,15);
        // B frag (16-bit 32x16 layout): lanes0-15 K 0..15, lanes16-31 K 16..31 (contiguous 32B)
        int kb2 = kb * 32 + half_sel * 16;
        v8h blo = *(const v8h*)(brow + kb2);
        v8h bhi = *(const v8h*)(brow + kb2 + 8);
        v16h bm = __builtin_shufflevector(blo, bhi, 0,1,2,3,4,5,6,7,8,9,10,11,12,13,14,15);
        acc = __builtin_amdgcn_wmma_f32_16x16x32_f16(
            /*neg_a=*/false, a, /*neg_b=*/false, bm,
            /*c_mod=*/(short)0, acc, /*reuse_a=*/false, /*reuse_b=*/false);
    }

    // ---- epilogue: C/D layout VGPR r -> M = r + 8*half_sel, N = ln16 ----
    int   n  = nb * 16 + ln16;
    float bn = bias[n];
    #pragma unroll
    for (int r = 0; r < 8; ++r) {
        int vt = r + half_sel * 8;
        int v  = v0 + vt;
        if (v < NVERT) {
            float m = smask[b * TILE_V + vt] ? 1.0f : 0.0f;
            out[((size_t)b * C_OUT + n) * NVERT + v] = (acc[r] + bn) * m;
        }
    }
}

extern "C" void kernel_launch(void* const* d_in, const int* in_sizes, int n_in,
                              void* d_out, int out_size, void* d_ws, size_t ws_size,
                              hipStream_t stream) {
    (void)in_sizes; (void)n_in; (void)out_size; (void)ws_size;
    const float* x    = (const float*)d_in[0];
    const int*   idx  = (const int*)  d_in[1];
    const float* itp  = (const float*)d_in[2];
    const float* w    = (const float*)d_in[3];
    const float* bvec = (const float*)d_in[4];
    float*  out = (float*)d_out;
    half_t* wh  = (half_t*)d_ws;                 // 32*288 f16 = 18,432 B of scratch

    wcvt_kernel<<<(C_OUT * KD + 255) / 256, 256, 0, stream>>>(w, wh);
    int nblk = (NVERT + TILE_V - 1) / TILE_V;    // 2561 blocks
    sphereconv_kernel<<<nblk, 256, 0, stream>>>(x, idx, itp, wh, bvec, out);
}
